// SelfAttn_26139170963464
// MI455X (gfx1250) — matrix operand order repaired and checked by reading, not compiled, not run on hardware
//
#include <hip/hip_runtime.h>

// ---------------------------------------------------------------------------
// Self-attention (fp32 in/out) for MI455X / gfx1250, bf16 WMMA compute.
//   DIM=1024, HEADS=16, HEAD_DIM=64, B=2, N=2048
// Pipeline:
//   0a) convert x -> bf16                 (memory-bound pre-pass, done once)
//   0b) convert+transpose W -> bf16 [n][k]
//   1)  fused QKV GEMM: async-LDS double-buffered, v_wmma_f32_16x16x32_bf16,
//       outputs head-split bf16 (Q pre-scaled by 1/8, V transposed)
//   2)  flash attention: K/V tiles async-staged to LDS, online softmax,
//       WMMA for Q.K^T and P.V
// ---------------------------------------------------------------------------

#define DIMF      1024
#define NHEADS    16
#define HDIM      64
#define SEQ       2048
#define BATCH     2
#define NTOK      (BATCH * SEQ)     // 4096
#define ATT_SCALE 0.125f            // HEAD_DIM^-0.5 (power of two: exact in bf16)

#define USE_ASYNC_LDS 1             // global_load_async_to_lds_b128 staging

typedef __attribute__((ext_vector_type(16))) __bf16         v16bf;
typedef __attribute__((ext_vector_type(8)))  float          v8f;
typedef __attribute__((ext_vector_type(4)))  unsigned int   v4u;
typedef __attribute__((ext_vector_type(4)))  float          v4f;
typedef __attribute__((ext_vector_type(4)))  unsigned short v4us;

union Frag16 {
    v16bf v;
    v4u   q[2];
};

__device__ __forceinline__ unsigned short f2bf(float f) {
    unsigned int u = __float_as_uint(f);
    u += 0x7FFFu + ((u >> 16) & 1u);     // round-to-nearest-even
    return (unsigned short)(u >> 16);
}

__device__ __forceinline__ v8f wmma_bf16(v16bf a, v16bf b, v8f c) {
    return __builtin_amdgcn_wmma_f32_16x16x32_bf16(false, a, false, b, (short)0, c,
                                                   false, false);
}

#if USE_ASYNC_LDS
// Per-lane 16B async copy global -> LDS (ASYNCcnt).  The VGPR dest operand is
// the LDS byte address = low 32 bits of the generic pointer to __shared__.
__device__ __forceinline__ void async_cp16(unsigned lds_byte_addr, const void* gaddr) {
    asm volatile("global_load_async_to_lds_b128 %0, %1, off"
                 :: "v"(lds_byte_addr), "v"(gaddr) : "memory");
}
__device__ __forceinline__ void async_wait_all() {
    asm volatile("s_wait_asynccnt 0" ::: "memory");
}
__device__ __forceinline__ unsigned lds_addr(const void* p) {
    return (unsigned)(size_t)p;          // generic LDS ptr low 32 bits = LDS offset
}
#endif

// ---------------------------------------------------------------------------
// Kernel 0a: x (fp32) -> xbf (bf16), 4 elements/thread.
// ---------------------------------------------------------------------------
__global__ __launch_bounds__(256)
void cvt_x_kernel(const float* __restrict__ x, unsigned short* __restrict__ xbf)
{
    size_t i = ((size_t)blockIdx.x * 256 + threadIdx.x) * 4;
    v4f f = *reinterpret_cast<const v4f*>(&x[i]);
    v4us h;
    h.x = f2bf(f.x); h.y = f2bf(f.y); h.z = f2bf(f.z); h.w = f2bf(f.w);
    *reinterpret_cast<v4us*>(&xbf[i]) = h;
}

// ---------------------------------------------------------------------------
// Kernel 0b: W[k][n] (fp32) -> Wt[z][n][k] (bf16), 64x64 LDS tile transpose.
// ---------------------------------------------------------------------------
__global__ __launch_bounds__(256)
void cvt_wt_kernel(const float* __restrict__ Wq, const float* __restrict__ Wk,
                   const float* __restrict__ Wv, unsigned short* __restrict__ Wt)
{
    __shared__ unsigned short tile[64 * 68];       // [n][k], stride 68 (pad)

    const int z = blockIdx.z;
    const float* W = (z == 0) ? Wq : (z == 1) ? Wk : Wv;
    const int kBase = blockIdx.y * 64;
    const int nBase = blockIdx.x * 64;
    const int tid = threadIdx.x;

#pragma unroll
    for (int it = 0; it < 4; ++it) {               // 1024 chunks of 4 floats
        int i  = tid + it * 256;
        int kl = i >> 4;
        int c4 = (i & 15) * 4;
        v4f f = *reinterpret_cast<const v4f*>(&W[(size_t)(kBase + kl) * DIMF + nBase + c4]);
#pragma unroll
        for (int j = 0; j < 4; ++j)
            tile[(c4 + j) * 68 + kl] = f2bf(f[j]);
    }
    __syncthreads();

    unsigned short* dst = Wt + (size_t)z * DIMF * DIMF;
#pragma unroll
    for (int it = 0; it < 4; ++it) {               // 1024 chunks of 4 bf16
        int i  = tid + it * 256;
        int nl = i >> 4;
        int c4 = (i & 15) * 4;
        v4us h = *reinterpret_cast<const v4us*>(&tile[nl * 68 + c4]);
        *reinterpret_cast<v4us*>(&dst[(size_t)(nBase + nl) * DIMF + kBase + c4]) = h;
    }
}

// ---------------------------------------------------------------------------
// Kernel 1: C[4096,1024] = xbf @ W + bias  (W pre-transposed bf16 [n][k]).
// Block tile 128x64, 8 waves x (16x64 strip), K-step 64, double-buffered
// async LDS staging.  Outputs:
//   z=0: Q[b,h,n,d] * ATT_SCALE   z=1: K[b,h,n,d]   z=2: Vt[b,h,d,n]
// ---------------------------------------------------------------------------
__global__ __launch_bounds__(256)
void qkv_gemm_kernel(const unsigned short* __restrict__ xbf,
                     const unsigned short* __restrict__ Wt,
                     const float* __restrict__ bq, const float* __restrict__ bk,
                     const float* __restrict__ bv,
                     unsigned short* __restrict__ Qb,
                     unsigned short* __restrict__ Kb,
                     unsigned short* __restrict__ Vt)
{
    __shared__ unsigned short sA[2][128 * 64];     // 2 x 16 KB
    __shared__ unsigned short sB[2][64 * 64];      // 2 x  8 KB

    const int z = blockIdx.z;
    const unsigned short* W = Wt + (size_t)z * DIMF * DIMF;
    const float* bias = (z == 0) ? bq : (z == 1) ? bk : bv;

    const int tid  = threadIdx.x;
    const int lane = tid & 31;
    const int wave = tid >> 5;
    const int hi   = lane >> 4;
    const int lr   = lane & 15;

    const int rowBase = blockIdx.y * 128;
    const int nBase   = blockIdx.x * 64;

    auto stage = [&](int buf, int k0) {
        // A tile: 128 rows x 64 bf16 = 1024 x 16B chunks (4/thread)
#pragma unroll
        for (int it = 0; it < 4; ++it) {
            int i = tid + it * 256;
            int r = i >> 3, c = i & 7;
            const unsigned short* g = &xbf[(size_t)(rowBase + r) * DIMF + k0 + c * 8];
#if USE_ASYNC_LDS
            async_cp16(lds_addr(&sA[buf][i * 8]), g);
#else
            *reinterpret_cast<v4u*>(&sA[buf][i * 8]) = *reinterpret_cast<const v4u*>(g);
#endif
        }
        // B tile: 64 rows(n) x 64 bf16 = 512 x 16B chunks (2/thread)
#pragma unroll
        for (int it = 0; it < 2; ++it) {
            int i = tid + it * 256;
            int n = i >> 3, c = i & 7;
            const unsigned short* g = &W[(size_t)(nBase + n) * DIMF + k0 + c * 8];
#if USE_ASYNC_LDS
            async_cp16(lds_addr(&sB[buf][i * 8]), g);
#else
            *reinterpret_cast<v4u*>(&sB[buf][i * 8]) = *reinterpret_cast<const v4u*>(g);
#endif
        }
    };

    v8f acc[4] = {};

    stage(0, 0);
#if USE_ASYNC_LDS
    async_wait_all();
#endif
    __syncthreads();

    for (int step = 0; step < DIMF / 64; ++step) {
        const int cur = step & 1;
        if (step + 1 < DIMF / 64)
            stage(cur ^ 1, (step + 1) * 64);       // overlap with WMMAs below

#pragma unroll
        for (int s = 0; s < 2; ++s) {              // K sub-steps of 32
            Frag16 a;
            int aoff = (wave * 16 + lr) * 64 + s * 32 + hi * 8;
            a.q[0] = *reinterpret_cast<const v4u*>(&sA[cur][aoff]);
            a.q[1] = *reinterpret_cast<const v4u*>(&sA[cur][aoff + 16]);
#pragma unroll
            for (int t = 0; t < 4; ++t) {
                Frag16 b;
                int boff = (t * 16 + lr) * 64 + s * 32 + hi * 8;
                b.q[0] = *reinterpret_cast<const v4u*>(&sB[cur][boff]);
                b.q[1] = *reinterpret_cast<const v4u*>(&sB[cur][boff + 16]);
                acc[t] = wmma_bf16(a.v, b.v, acc[t]);
            }
        }
#if USE_ASYNC_LDS
        async_wait_all();
#endif
        __syncthreads();
    }

    const float oscale = (z == 0) ? ATT_SCALE : 1.0f;   // fold softmax scale into Q
#pragma unroll
    for (int t = 0; t < 4; ++t) {
        int col = nBase + t * 16 + lr;
        int h = col >> 6, d = col & 63;
        float bcol = bias[col];
#pragma unroll
        for (int i = 0; i < 8; ++i) {
            int rowg = rowBase + wave * 16 + hi * 8 + i;
            int bb = rowg >> 11, np = rowg & 2047;
            unsigned short val = f2bf((acc[t][i] + bcol) * oscale);
            size_t bh = (size_t)bb * NHEADS + h;
            if (z == 0)      Qb[(bh * SEQ + np) * HDIM + d] = val;
            else if (z == 1) Kb[(bh * SEQ + np) * HDIM + d] = val;
            else             Vt[(bh * HDIM + d) * SEQ + np] = val;
        }
    }
}

// ---------------------------------------------------------------------------
// Kernel 2: flash attention.  grid = (32 qblocks of 64, 32 b*h), 4 waves.
// 64-key tiles: K/V async-staged to LDS once per block (4x traffic saving),
// 8 WMMAs for S, online softmax, 8 WMMAs for P.V.  Q already carries SCALE.
// ---------------------------------------------------------------------------
__global__ __launch_bounds__(128)
void flash_attn_kernel(const unsigned short* __restrict__ Qb,
                       const unsigned short* __restrict__ Kb,
                       const unsigned short* __restrict__ Vt,
                       float* __restrict__ out)
{
    __shared__ unsigned short sK[2][64 * 64];      // [key][d]   2 x 8 KB
    __shared__ unsigned short sV[2][64 * 64];      // [d][key]   2 x 8 KB
    __shared__ unsigned short sP[4][16 * 64];      // per-wave P tile, 8 KB

    const int tid  = threadIdx.x;
    const int lane = tid & 31;
    const int wave = tid >> 5;
    const int hi   = lane >> 4;
    const int lr   = lane & 15;

    const int bh = blockIdx.y;
    const int b  = bh >> 4;
    const int h  = bh & 15;
    const int qrow0 = blockIdx.x * 64 + wave * 16;

    const unsigned short* Qh = Qb + (size_t)bh * SEQ * HDIM;
    const unsigned short* Kh = Kb + (size_t)bh * SEQ * HDIM;
    const unsigned short* Vh = Vt + (size_t)bh * HDIM * SEQ;

    auto stage = [&](int buf, int kt) {
        // K tile: 64 keys x 64 d  = 512 x 16B chunks (4/thread)
#pragma unroll
        for (int it = 0; it < 4; ++it) {
            int i = tid + it * 128;
            int r = i >> 3, c = i & 7;
            const unsigned short* g = &Kh[(size_t)(kt + r) * HDIM + c * 8];
#if USE_ASYNC_LDS
            async_cp16(lds_addr(&sK[buf][i * 8]), g);
#else
            *reinterpret_cast<v4u*>(&sK[buf][i * 8]) = *reinterpret_cast<const v4u*>(g);
#endif
        }
        // V tile: 64 d x 64 keys = 512 x 16B chunks (4/thread)
#pragma unroll
        for (int it = 0; it < 4; ++it) {
            int i = tid + it * 128;
            int d = i >> 3, c = i & 7;
            const unsigned short* g = &Vh[(size_t)d * SEQ + kt + c * 8];
#if USE_ASYNC_LDS
            async_cp16(lds_addr(&sV[buf][i * 8]), g);
#else
            *reinterpret_cast<v4u*>(&sV[buf][i * 8]) = *reinterpret_cast<const v4u*>(g);
#endif
        }
    };

    // Q fragments (d halves), resident in registers; Q is pre-scaled by 1/8.
    Frag16 qf[2];
#pragma unroll
    for (int s = 0; s < 2; ++s) {
        int off = (qrow0 + lr) * HDIM + s * 32 + hi * 8;
        qf[s].q[0] = *reinterpret_cast<const v4u*>(&Qh[off]);
        qf[s].q[1] = *reinterpret_cast<const v4u*>(&Qh[off + 16]);
    }

    float m[8], l[8];
#pragma unroll
    for (int i = 0; i < 8; ++i) { m[i] = -1e30f; l[i] = 0.0f; }
    v8f acc[4] = {};

    stage(0, 0);
#if USE_ASYNC_LDS
    async_wait_all();
#endif
    __syncthreads();

    for (int step = 0; step < SEQ / 64; ++step) {
        const int cur = step & 1;
        if (step + 1 < SEQ / 64)
            stage(cur ^ 1, (step + 1) * 64);

        // ---- S = Q.K^T : 4 col-tiles x 2 d-halves ----
        v8f sc4[4] = {};
#pragma unroll
        for (int ct = 0; ct < 4; ++ct) {
#pragma unroll
            for (int s = 0; s < 2; ++s) {
                Frag16 kf;
                int off = (ct * 16 + lr) * 64 + s * 32 + hi * 8;
                kf.q[0] = *reinterpret_cast<const v4u*>(&sK[cur][off]);
                kf.q[1] = *reinterpret_cast<const v4u*>(&sK[cur][off + 16]);
                sc4[ct] = wmma_bf16(qf[s].v, kf.v, sc4[ct]);
            }
        }

        // ---- online softmax over 64 keys ----
#pragma unroll
        for (int i = 0; i < 8; ++i) {
            float a0 = sc4[0][i], a1 = sc4[1][i], a2 = sc4[2][i], a3 = sc4[3][i];
            float cur_m = fmaxf(fmaxf(a0, a1), fmaxf(a2, a3));
#pragma unroll
            for (int d = 1; d < 16; d <<= 1) cur_m = fmaxf(cur_m, __shfl_xor(cur_m, d, 16));
            float mn = fmaxf(m[i], cur_m);
            float sc = __expf(m[i] - mn);
            float p0 = __expf(a0 - mn), p1 = __expf(a1 - mn);
            float p2 = __expf(a2 - mn), p3 = __expf(a3 - mn);
            float rs = (p0 + p1) + (p2 + p3);
#pragma unroll
            for (int d = 1; d < 16; d <<= 1) rs += __shfl_xor(rs, d, 16);
            l[i] = l[i] * sc + rs;
            m[i] = mn;
#pragma unroll
            for (int t = 0; t < 4; ++t) acc[t][i] *= sc;

            int pr = (hi * 8 + i) * 64 + lr;
            sP[wave][pr]      = f2bf(p0);
            sP[wave][pr + 16] = f2bf(p1);
            sP[wave][pr + 32] = f2bf(p2);
            sP[wave][pr + 48] = f2bf(p3);
        }

        // ---- acc += P.V  (sP is wave-private; same-wave LDS is in-order) ----
#pragma unroll
        for (int s = 0; s < 2; ++s) {              // key halves 0..31 / 32..63
            Frag16 pf;
            int poff = lr * 64 + s * 32 + hi * 8;
            pf.q[0] = *reinterpret_cast<const v4u*>(&sP[wave][poff]);
            pf.q[1] = *reinterpret_cast<const v4u*>(&sP[wave][poff + 16]);
#pragma unroll
            for (int t = 0; t < 4; ++t) {
                Frag16 vf;
                int off = (t * 16 + lr) * 64 + s * 32 + hi * 8;
                vf.q[0] = *reinterpret_cast<const v4u*>(&sV[cur][off]);
                vf.q[1] = *reinterpret_cast<const v4u*>(&sV[cur][off + 16]);
                acc[t] = wmma_bf16(pf.v, vf.v, acc[t]);
            }
        }
#if USE_ASYNC_LDS
        async_wait_all();
#endif
        __syncthreads();
    }

    // ---- normalize and store fp32 [b, n, 1024] ----
#pragma unroll
    for (int t = 0; t < 4; ++t) {
#pragma unroll
        for (int i = 0; i < 8; ++i) {
            int np = qrow0 + hi * 8 + i;
            out[((size_t)b * SEQ + np) * DIMF + h * HDIM + t * 16 + lr] = acc[t][i] / l[i];
        }
    }
}

// ---------------------------------------------------------------------------
extern "C" void kernel_launch(void* const* d_in, const int* in_sizes, int n_in,
                              void* d_out, int out_size, void* d_ws, size_t ws_size,
                              hipStream_t stream)
{
    (void)in_sizes; (void)n_in; (void)out_size; (void)ws_size;

    const float* x  = (const float*)d_in[0];
    const float* Wq = (const float*)d_in[1];
    const float* bq = (const float*)d_in[2];
    const float* Wk = (const float*)d_in[3];
    const float* bk = (const float*)d_in[4];
    const float* Wv = (const float*)d_in[5];
    const float* bv = (const float*)d_in[6];

    // workspace (bf16): xbf 8MB | Wt 6MB | Q 8MB | K 8MB | Vt 8MB  = 38 MB
    const size_t qkv_elems = (size_t)BATCH * NHEADS * SEQ * HDIM;   // 4M
    unsigned short* xbf = (unsigned short*)d_ws;
    unsigned short* Wt  = xbf + (size_t)NTOK * DIMF;
    unsigned short* Qb  = Wt + (size_t)3 * DIMF * DIMF;
    unsigned short* Kb  = Qb + qkv_elems;
    unsigned short* Vt  = Kb + qkv_elems;

    cvt_x_kernel<<<dim3((NTOK * DIMF) / (256 * 4)), 256, 0, stream>>>(x, xbf);
    cvt_wt_kernel<<<dim3(16, 16, 3), 256, 0, stream>>>(Wq, Wk, Wv, Wt);
    qkv_gemm_kernel<<<dim3(16, 32, 3), 256, 0, stream>>>(
        xbf, Wt, bq, bk, bv, Qb, Kb, Vt);
    flash_attn_kernel<<<dim3(32, 32), 128, 0, stream>>>(
        Qb, Kb, Vt, (float*)d_out);
}
